// DampedMultidimEMACumsum_71116068487383
// MI455X (gfx1250) — matrix operand order repaired and checked by reading, not compiled, not run on hardware
//
#include <hip/hip_runtime.h>

// ---------------------------------------------------------------------------
// DampedMultidimEMACumsum for MI455X (gfx1250, wave32, WMMA)
// B=8 N=4096 C=512 H=4 ; out = concat(ema_fwd, ema_bwd) @ W.T + b
//
// - EMA implemented exactly as the reference's clamped cumsum (NOT a plain
//   recurrence: the +-60 clamp changes semantics for fast-decaying channels).
//   The scaled cumsum is linear -> 3-pass chunked scan (partials / exclusive
//   prefix / rescan).
// - Rescan emits Z pre-split into bf16 hi/lo planes so the GEMM inner loop is
//   pure loads + v_wmma_f32_16x16x32_bf16 (split-bf16: Zh*Wh + Zl*Wh + Zh*Wl,
//   ~fp32 accuracy at 3 WMMAs, 8x fewer matrix-op issues than f32 16x16x4).
// ---------------------------------------------------------------------------

typedef __attribute__((ext_vector_type(16))) __bf16 v16bf;
typedef __attribute__((ext_vector_type(8)))  float  v8f;
typedef unsigned short ushort_t;

#define B_    8
#define N_    4096
#define C_    512
#define H_    4
#define K2C   1024            // 2*C  (GEMM K)
#define M_    (B_ * N_)       // 32768 (GEMM M)
#define SCH   32              // scan chunks
#define LCH   128             // chunk length (SCH*LCH == N_)
#define CLAMP2 86.5616989f    // 60 * log2(e)

__device__ __forceinline__ ushort_t f2bf(float f) {          // RNE fp32->bf16
  unsigned u = __float_as_uint(f);
  u += 0x7fffu + ((u >> 16) & 1u);
  return (ushort_t)(u >> 16);
}
__device__ __forceinline__ float bf2f(ushort_t h) {
  return __uint_as_float(((unsigned)h) << 16);
}

// --------------------------- weight tables ---------------------------------
__global__ void k_weights(const float* __restrict__ al, const float* __restrict__ dl,
                          const float* __restrict__ bl,
                          float* __restrict__ Atab, float* __restrict__ Ltab) {
  int i = blockIdx.x * blockDim.x + threadIdx.x;
  if (i >= C_ * H_) return;
  float sa = 1.0f / (1.0f + __expf(-al[i]));
  float sd = 1.0f / (1.0f + __expf(-dl[i]));
  float sb = 1.0f / (1.0f + __expf(-bl[i]));
  Atab[i] = sa * sb;
  float r = 1.0f - sa * sd;
  r = fminf(fmaxf(r, 1e-4f), 1.0f - 1e-4f);
  Ltab[i] = -__log2f(r);                 // |log2 r| >= 0
}

// ---------------------- proj_w split into bf16 hi/lo -----------------------
__global__ void k_wsplit(const float* __restrict__ w,
                         ushort_t* __restrict__ whi, ushort_t* __restrict__ wlo) {
  int i = blockIdx.x * blockDim.x + threadIdx.x;
  if (i >= C_ * K2C) return;
  float f = w[i];
  ushort_t h = f2bf(f);
  whi[i] = h;
  wlo[i] = f2bf(f - bf2f(h));
}

// ------------------- pass 1: per-chunk partial cumsums ---------------------
__global__ void k_ema_partial(const float* __restrict__ x,
                              const float* __restrict__ Atab,
                              const float* __restrict__ Ltab,
                              float* __restrict__ Pf, float* __restrict__ Pb) {
  int tid = blockIdx.x * blockDim.x + threadIdx.x;   // (k*B + b)*C + c
  int c  = tid % C_;
  int t2 = tid / C_;
  int b  = t2 % B_;
  int k  = t2 / B_;
  float A[H_], L[H_];
#pragma unroll
  for (int h = 0; h < H_; ++h) { A[h] = Atab[c * H_ + h]; L[h] = Ltab[c * H_ + h]; }
  float aF[H_] = {0.f, 0.f, 0.f, 0.f};
  float aB[H_] = {0.f, 0.f, 0.f, 0.f};
  const float* xp = x + (size_t)b * N_ * C_ + c;
  int t0 = k * LCH;
  for (int i = 0; i < LCH; ++i) {
    int t = t0 + i;
    float tf = (float)t;
    float xf = xp[(size_t)t * C_];
    float xr = xp[(size_t)(N_ - 1 - t) * C_];
    __builtin_prefetch(xp + (size_t)(t + 8) * C_, 0, 1);          // streaming
    __builtin_prefetch(xp + (size_t)(N_ - 9 - t) * C_, 0, 1);
#pragma unroll
    for (int h = 0; h < H_; ++h) {
      float q = fminf(L[h] * tf, CLAMP2);
      float z = A[h] * exp2f(q);          // A * r_neg[t]
      aF[h] += z * xf;
      aB[h] += z * xr;
    }
  }
  size_t p = (size_t)tid * H_;
  *(float4*)(Pf + p) = make_float4(aF[0], aF[1], aF[2], aF[3]);
  *(float4*)(Pb + p) = make_float4(aB[0], aB[1], aB[2], aB[3]);
}

// ------------------- pass 2: exclusive prefix over chunks ------------------
__global__ void k_ema_prefix(float* __restrict__ Pf, float* __restrict__ Pb) {
  int tid = blockIdx.x * blockDim.x + threadIdx.x;   // (b*C+c)*H + h
  if (tid >= B_ * C_ * H_) return;
  float cf = 0.f, cb = 0.f;
  for (int k = 0; k < SCH; ++k) {
    size_t idx = (size_t)k * (B_ * C_ * H_) + tid;
    float tf = Pf[idx]; Pf[idx] = cf; cf += tf;
    float tb = Pb[idx]; Pb[idx] = cb; cb += tb;
  }
}

// ---- pass 3: rescan with carried prefix; emit Z split as bf16 hi/lo -------
__global__ void k_ema_scan(const float* __restrict__ x,
                           const float* __restrict__ Atab,
                           const float* __restrict__ Ltab,
                           const float* __restrict__ eta,
                           const float* __restrict__ Pf,
                           const float* __restrict__ Pb,
                           ushort_t* __restrict__ Zhi,
                           ushort_t* __restrict__ Zlo) {
  int tid = blockIdx.x * blockDim.x + threadIdx.x;   // (k*B + b)*C + c
  int c  = tid % C_;
  int t2 = tid / C_;
  int b  = t2 % B_;
  int k  = t2 / B_;
  float A[H_], L[H_], E[H_];
#pragma unroll
  for (int h = 0; h < H_; ++h) {
    A[h] = Atab[c * H_ + h]; L[h] = Ltab[c * H_ + h]; E[h] = eta[c * H_ + h];
  }
  float sF[H_], sB[H_];
  size_t p = (size_t)tid * H_;
  float4 vF = *(const float4*)(Pf + p);
  float4 vB = *(const float4*)(Pb + p);
  sF[0] = vF.x; sF[1] = vF.y; sF[2] = vF.z; sF[3] = vF.w;
  sB[0] = vB.x; sB[1] = vB.y; sB[2] = vB.z; sB[3] = vB.w;

  const float* xp = x + (size_t)b * N_ * C_ + c;
  ushort_t* zh = Zhi + (size_t)b * N_ * K2C;
  ushort_t* zl = Zlo + (size_t)b * N_ * K2C;
  int t0 = k * LCH;
  for (int i = 0; i < LCH; ++i) {
    int t = t0 + i;
    float tf = (float)t;
    float xf = xp[(size_t)t * C_];
    float xr = xp[(size_t)(N_ - 1 - t) * C_];
    __builtin_prefetch(xp + (size_t)(t + 8) * C_, 0, 1);
    __builtin_prefetch(xp + (size_t)(N_ - 9 - t) * C_, 0, 1);
    float yf = 0.f, yb = 0.f;
#pragma unroll
    for (int h = 0; h < H_; ++h) {
      float q  = fminf(L[h] * tf, CLAMP2);
      float rn = exp2f(q);                 // r_neg[t]
      float rp = exp2f(-q);                // r_pos[t]
      sF[h] += A[h] * rn * xf;
      sB[h] += A[h] * rn * xr;
      yf += E[h] * sF[h] * rp;
      yb += E[h] * sB[h] * rp;
    }
    // forward result -> cols [0,512), backward -> cols [512,1024)
    size_t iF = (size_t)t * K2C + c;
    size_t iB = (size_t)(N_ - 1 - t) * K2C + C_ + c;
    ushort_t hf = f2bf(yf);
    zh[iF] = hf; zl[iF] = f2bf(yf - bf2f(hf));
    ushort_t hb = f2bf(yb);
    zh[iB] = hb; zl[iB] = f2bf(yb - bf2f(hb));
  }
}

// ----------------------------- GEMM (WMMA) ---------------------------------
// out[m,j] = sum_k Z[m,k] * W[j,k] + bias[j]
// Block = 256 threads (8 waves). Block owns 16 M-rows; wave w owns cols
// [64w, 64w+64) as 4 WMMA N-tiles. All operands pre-split bf16 -> inner loop
// is pure global_load_b128 + v_wmma (A slab shared by 8 waves via WGP$).
__global__ void __launch_bounds__(256) k_gemm(const ushort_t* __restrict__ Zhi,
                                              const ushort_t* __restrict__ Zlo,
                                              const ushort_t* __restrict__ Whi,
                                              const ushort_t* __restrict__ Wlo,
                                              const float* __restrict__ bias,
                                              float* __restrict__ out) {
  const int lane  = threadIdx.x & 31;
  const int wv    = threadIdx.x >> 5;      // 0..7
  const int mBase = blockIdx.x * 16;
  const int rowA  = lane & 15;             // A row / B-D column within tile
  const int hsel  = lane >> 4;             // lane half: 0/1
  const int s0    = hsel * 8;              // A-frag K sub-offset (interleaved K)
  const int bk16  = hsel * 8;              // B-frag K sub-offset (interleaved K)
  const int n0    = wv * 64;

  const v8f z8 = {0.f, 0.f, 0.f, 0.f, 0.f, 0.f, 0.f, 0.f};
  v8f acc[4] = {z8, z8, z8, z8};

  union Frag  { v16bf v; ushort_t u[16]; };
  union FragF { v8f v; float f[8]; };

  // per-lane row pointers (A layout: lane<16 -> K {0..7,16..23}, else +8)
  const ushort_t* zrh = Zhi + (size_t)(mBase + rowA) * K2C + s0;
  const ushort_t* zrl = Zlo + (size_t)(mBase + rowA) * K2C + s0;

  for (int kb = 0; kb < K2C; kb += 32) {
    Frag ahi, alo;
    *(uint4*)&ahi.u[0] = *(const uint4*)(zrh + kb);
    *(uint4*)&ahi.u[8] = *(const uint4*)(zrh + kb + 16);
    *(uint4*)&alo.u[0] = *(const uint4*)(zrl + kb);
    *(uint4*)&alo.u[8] = *(const uint4*)(zrl + kb + 16);

#pragma unroll
    for (int t = 0; t < 4; ++t) {
      const int n = n0 + 16 * t + rowA;    // output column this lane feeds
      const ushort_t* ph = Whi + (size_t)n * K2C + kb + bk16;
      const ushort_t* pl = Wlo + (size_t)n * K2C + kb + bk16;
      Frag bhi, blo;
      *(uint4*)&bhi.u[0] = *(const uint4*)(ph);      // K interleaved: {0..7,16..23} (+8)
      *(uint4*)&bhi.u[8] = *(const uint4*)(ph + 16);
      *(uint4*)&blo.u[0] = *(const uint4*)(pl);
      *(uint4*)&blo.u[8] = *(const uint4*)(pl + 16);
      acc[t] = __builtin_amdgcn_wmma_f32_16x16x32_bf16(false, ahi.v, false, bhi.v,
                                                       (short)0, acc[t], false, false);
      acc[t] = __builtin_amdgcn_wmma_f32_16x16x32_bf16(false, alo.v, false, bhi.v,
                                                       (short)0, acc[t], false, false);
      acc[t] = __builtin_amdgcn_wmma_f32_16x16x32_bf16(false, ahi.v, false, blo.v,
                                                       (short)0, acc[t], false, false);
    }
  }

#pragma unroll
  for (int t = 0; t < 4; ++t) {
    const int col = n0 + 16 * t + rowA;
    const float bv = bias[col];
    FragF a; a.v = acc[t];
#pragma unroll
    for (int g = 0; g < 8; ++g) {
      const int r = mBase + g + hsel * 8;  // D layout: VGPR g -> M=g / g+8
      out[(size_t)r * C_ + col] = a.f[g] + bv;
    }
  }
}

// ---------------------------------------------------------------------------
extern "C" void kernel_launch(void* const* d_in, const int* in_sizes, int n_in,
                              void* d_out, int out_size, void* d_ws, size_t ws_size,
                              hipStream_t stream) {
  const float* x   = (const float*)d_in[0];
  const float* al  = (const float*)d_in[1];
  const float* dl  = (const float*)d_in[2];
  const float* bl  = (const float*)d_in[3];
  const float* eta = (const float*)d_in[4];
  const float* pw  = (const float*)d_in[5];
  const float* pb  = (const float*)d_in[6];
  float* out = (float*)d_out;

  char* w = (char*)d_ws;
  float*    Atab = (float*)(w);                                        //   8 KB
  float*    Ltab = (float*)(w + (8u << 10));                           //   8 KB
  ushort_t* Whi  = (ushort_t*)(w + (16u << 10));                       //   1 MB
  ushort_t* Wlo  = (ushort_t*)(w + (16u << 10) + (1u << 20));          //   1 MB
  float*    Pf   = (float*)(w + (16u << 10) + (2u << 20));             //   2 MB
  float*    Pb   = (float*)(w + (16u << 10) + (4u << 20));             //   2 MB
  ushort_t* Zhi  = (ushort_t*)(w + (16u << 10) + (6u << 20));          //  64 MB
  ushort_t* Zlo  = (ushort_t*)(w + (16u << 10) + (70u << 20));         //  64 MB

  k_weights<<<(C_ * H_ + 255) / 256, 256, 0, stream>>>(al, dl, bl, Atab, Ltab);
  k_wsplit<<<(C_ * K2C + 255) / 256, 256, 0, stream>>>(pw, Whi, Wlo);
  k_ema_partial<<<(SCH * B_ * C_) / 256, 256, 0, stream>>>(x, Atab, Ltab, Pf, Pb);
  k_ema_prefix<<<(B_ * C_ * H_) / 256, 256, 0, stream>>>(Pf, Pb);
  k_ema_scan<<<(SCH * B_ * C_) / 256, 256, 0, stream>>>(x, Atab, Ltab, eta, Pf, Pb,
                                                        Zhi, Zlo);
  k_gemm<<<M_ / 16, 256, 0, stream>>>(Zhi, Zlo, Whi, Wlo, pb, out);
}
